// Parallel_LSTM_CRF_38405597561674
// MI455X (gfx1250) — matrix operand order repaired and checked by reading, not compiled, not run on hardware
//
#include <hip/hip_runtime.h>

// ---------------- problem constants (from reference) ----------------
#define S_LEN   2048
#define E_DIM   512
#define H_W     1000      // word lstm hidden
#define H_C     200       // char lstm hidden
#define OC_DIM  40
#define DC_DIM  40
#define L_CH    20
#define T_TAG   19
#define START_IX 17
#define STOP_IX  18
#define NEGVAL  -10000.0f
#define CAT_DIM 2400      // 2*H_W + 2*H_C
#define HALF_H  500       // H_W/2
#define HALF_HP 512       // padded
#define TPAD    32        // padded tagset
#define OC_PAD  64        // padded char-feature K

typedef __attribute__((ext_vector_type(16))) __bf16 v16bf;
typedef __attribute__((ext_vector_type(8)))  float  v8f;

__device__ __forceinline__ unsigned short f2bf(float f) {
  unsigned int u = __float_as_uint(f);
  u += 0x7FFFu + ((u >> 16) & 1u);          // round-to-nearest-even
  return (unsigned short)(u >> 16);
}
__device__ __forceinline__ float sigmoidf_(float x) { return 1.0f / (1.0f + __expf(-x)); }

// ---------------- generic conversion / padding kernels ----------------
__global__ void k_convert_bf16(const float* __restrict__ src, unsigned short* __restrict__ dst, int n) {
  int i = blockIdx.x * blockDim.x + threadIdx.x;
  if (i < n) dst[i] = f2bf(src[i]);
}

__global__ void k_convert_pad_bf16(const float* __restrict__ src, unsigned short* __restrict__ dst,
                                   int src_rows, int src_cols, int dst_rows, int dst_cols) {
  int i = blockIdx.x * blockDim.x + threadIdx.x;
  int n = dst_rows * dst_cols;
  if (i >= n) return;
  int r = i / dst_cols, c = i % dst_cols;
  float v = (r < src_rows && c < src_cols) ? src[r * src_cols + c] : 0.0f;
  dst[i] = f2bf(v);
}

__global__ void k_pad_f32(const float* __restrict__ src, float* __restrict__ dst, int n_src, int n_dst) {
  int i = blockIdx.x * blockDim.x + threadIdx.x;
  if (i < n_dst) dst[i] = (i < n_src) ? src[i] : 0.0f;
}

__global__ void k_gather_emb(const float* __restrict__ emb, const int* __restrict__ sent,
                             unsigned short* __restrict__ dst) {
  int i = blockIdx.x * blockDim.x + threadIdx.x;
  if (i >= S_LEN * E_DIM) return;
  int s = i / E_DIM, e = i % E_DIM;
  dst[i] = f2bf(emb[(size_t)sent[s] * E_DIM + e]);
}

// ---------------- char CNN: conv(3xDC, pad 2) + maxpool over time ----------------
__global__ void k_char_cnn(const int* __restrict__ chars2, const float* __restrict__ char_emb,
                           const float* __restrict__ conv_w, const float* __restrict__ conv_b,
                           float* __restrict__ cfeat) {
  int idx = blockIdx.x * blockDim.x + threadIdx.x;
  if (idx >= S_LEN * OC_DIM) return;
  int s = idx / OC_DIM, oc = idx % OC_DIM;
  const float* w = conv_w + oc * 3 * DC_DIM;
  float best = -3.4e38f;
  for (int i = 0; i < L_CH + 2; ++i) {
    float acc = 0.0f;
    #pragma unroll
    for (int kh = 0; kh < 3; ++kh) {
      int r = i + kh - 2;
      if (r < 0 || r >= L_CH) continue;
      const float* e  = char_emb + chars2[s * L_CH + r] * DC_DIM;
      const float* wk = w + kh * DC_DIM;
      for (int d = 0; d < DC_DIM; ++d) acc += e[d] * wk[d];
    }
    best = fmaxf(best, acc);
  }
  cfeat[idx] = best + conv_b[oc];
}

// ---------------- char highway (tiny 40x40 mats, VALU) ----------------
__global__ void k_char_hw_flow(const float* __restrict__ x, const float* __restrict__ fc1w,
                               const float* __restrict__ fc1b, const float* __restrict__ gw,
                               const float* __restrict__ gb, float* __restrict__ flow) {
  int idx = blockIdx.x * blockDim.x + threadIdx.x;
  if (idx >= S_LEN * OC_DIM) return;
  int s = idx / OC_DIM, j = idx % OC_DIM;
  const float* xr = x + s * OC_DIM;
  float a = fc1b[j], t = gb[j];
  for (int d = 0; d < OC_DIM; ++d) { a += fc1w[j * OC_DIM + d] * xr[d]; t += gw[j * OC_DIM + d] * xr[d]; }
  float tt = sigmoidf_(t);
  flow[idx] = tanhf(a) * tt + xr[j] * (1.0f - tt);
}

__global__ void k_char_hw_lin(const float* __restrict__ flow, const float* __restrict__ lw,
                              const float* __restrict__ lb, float* __restrict__ out) {
  int idx = blockIdx.x * blockDim.x + threadIdx.x;
  if (idx >= S_LEN * OC_DIM) return;
  int s = idx / OC_DIM, j = idx % OC_DIM;
  const float* fr = flow + s * OC_DIM;
  float a = lb[j];
  for (int d = 0; d < OC_DIM; ++d) a += lw[j * OC_DIM + d] * fr[d];
  out[idx] = a;
}

// ---------------- WMMA bf16 GEMM:  C[M,N] = A[M,K] @ W[N,K]^T (+bias[N]) ----------------
// grid = (N/16, M/256), block = 128 (4 waves). All 4 waves share one 16-col B strip;
// wave 0 streams the next 16x32 B chunk into LDS with global_load_async_to_lds_b128
// (double-buffered, ASYNCcnt), every wave register-blocks 4 M-tiles -> 4 WMMAs per B frag.
union FragAB { v16bf v; uint4 q[2]; };

__global__ void __launch_bounds__(128)
k_gemm_bf16_wmma(const unsigned short* __restrict__ A,
                 const unsigned short* __restrict__ W,
                 const float* __restrict__ bias,
                 float* __restrict__ C, int N, int K) {
  __shared__ unsigned short b_sh[2][16 * 32];   // two 1KB buffers: 16 rows x 32 bf16

  int wave = threadIdx.x >> 5;
  int lane = threadIdx.x & 31;
  int half = lane >> 4;          // 0: lanes 0-15, 1: lanes 16-31
  int r16  = lane & 15;
  int n0    = blockIdx.x * 16;
  int mbase = blockIdx.y * 256 + wave * 64;

  // --- async B-copy addressing (wave 0 issues for all) ---
  // lane -> row r16, 16B segment `half`; two b128 issues cover the full 16x32 chunk.
  const unsigned short* gB = W + (size_t)(n0 + r16) * K + half * 8;
  // low 32 bits of a generic LDS pointer == LDS byte offset
  unsigned ldsB    = (unsigned)(unsigned long long)&b_sh[0][r16 * 32 + half * 8];
  unsigned ldsBufD = (unsigned)((unsigned long long)&b_sh[1][0] - (unsigned long long)&b_sh[0][0]);

  if (wave == 0) {
    asm volatile("global_load_async_to_lds_b128 %0, %1, off" :: "v"(ldsB), "v"(gB) : "memory");
    asm volatile("global_load_async_to_lds_b128 %0, %1, off" :: "v"(ldsB + 32u), "v"(gB + 16) : "memory");
    asm volatile("s_wait_asynccnt 0x0" ::: "memory");
  }
  __syncthreads();

  v8f acc0 = {0.f,0.f,0.f,0.f,0.f,0.f,0.f,0.f};
  v8f acc1 = acc0, acc2 = acc0, acc3 = acc0;

  // A row base pointers for the 4 M-tiles (ISA 7.12.2 A layout: K {0..7,16..23}/{8..15,24..31})
  const unsigned short* a0 = A + (size_t)(mbase +  0 + r16) * K + half * 8;
  const unsigned short* a1 = A + (size_t)(mbase + 16 + r16) * K + half * 8;
  const unsigned short* a2 = A + (size_t)(mbase + 32 + r16) * K + half * 8;
  const unsigned short* a3 = A + (size_t)(mbase + 48 + r16) * K + half * 8;

  for (int k0 = 0; k0 < K; k0 += 32) {
    int buf = (k0 >> 5) & 1;
    bool more = (k0 + 32) < K;
    if (wave == 0 && more) {       // prefetch next B chunk into the other LDS buffer
      const unsigned short* g = gB + (k0 + 32);
      unsigned l = ldsB + (unsigned)(buf ^ 1) * ldsBufD;
      asm volatile("global_load_async_to_lds_b128 %0, %1, off" :: "v"(l), "v"(g) : "memory");
      asm volatile("global_load_async_to_lds_b128 %0, %1, off" :: "v"(l + 32u), "v"(g + 16) : "memory");
    }

    // B 32x16 fragment from LDS (lanes 0-15: K 0..7,16..23, lanes 16-31: K 8..15,24..31)
    const unsigned short* bp = &b_sh[buf][r16 * 32 + half * 8];
    FragAB b;
    b.q[0] = *(const uint4*)(bp);
    b.q[1] = *(const uint4*)(bp + 16);

    FragAB a;
    a.q[0] = *(const uint4*)(a0 + k0); a.q[1] = *(const uint4*)(a0 + k0 + 16);
    acc0 = __builtin_amdgcn_wmma_f32_16x16x32_bf16(false, a.v, false, b.v, (short)0, acc0, false, false);
    a.q[0] = *(const uint4*)(a1 + k0); a.q[1] = *(const uint4*)(a1 + k0 + 16);
    acc1 = __builtin_amdgcn_wmma_f32_16x16x32_bf16(false, a.v, false, b.v, (short)0, acc1, false, false);
    a.q[0] = *(const uint4*)(a2 + k0); a.q[1] = *(const uint4*)(a2 + k0 + 16);
    acc2 = __builtin_amdgcn_wmma_f32_16x16x32_bf16(false, a.v, false, b.v, (short)0, acc2, false, false);
    a.q[0] = *(const uint4*)(a3 + k0); a.q[1] = *(const uint4*)(a3 + k0 + 16);
    acc3 = __builtin_amdgcn_wmma_f32_16x16x32_bf16(false, a.v, false, b.v, (short)0, acc3, false, false);

    if (more) {                    // keep the A streams flowing toward L2/L0
      __builtin_prefetch(a0 + k0 + 32, 0, 1);
      __builtin_prefetch(a1 + k0 + 32, 0, 1);
      __builtin_prefetch(a2 + k0 + 32, 0, 1);
      __builtin_prefetch(a3 + k0 + 32, 0, 1);
      if (wave == 0)
        asm volatile("s_wait_asynccnt 0x0" ::: "memory");
    }
    __syncthreads();
  }

  int col = n0 + r16;
  float badd = bias ? bias[col] : 0.0f;
  // C/D layout: VGPR v holds M = v (lanes 0-15) / v+8 (lanes 16-31), N = lane&15
  #pragma unroll
  for (int v = 0; v < 8; ++v) {
    int r = v + half * 8;
    C[(size_t)(mbase +  0 + r) * N + col] = acc0[v] + badd;
    C[(size_t)(mbase + 16 + r) * N + col] = acc1[v] + badd;
    C[(size_t)(mbase + 32 + r) * N + col] = acc2[v] + badd;
    C[(size_t)(mbase + 48 + r) * N + col] = acc3[v] + badd;
  }
}

// ---------------- big highway elementwise fuse ----------------
__global__ void k_hw_fuse(const float* __restrict__ hwh, const float* __restrict__ hwt,
                          const float* __restrict__ x, unsigned short* __restrict__ flow_bf, int n) {
  int i = blockIdx.x * blockDim.x + threadIdx.x;
  if (i >= n) return;
  float t = sigmoidf_(hwt[i]);
  flow_bf[i] = f2bf(tanhf(hwh[i]) * t + x[i] * (1.0f - t));
}

// ---------------- sequential LSTM scan (one workgroup per direction) ----------------
__global__ void k_lstm_scan(const float* __restrict__ gx_f, const float* __restrict__ gx_b,
                            const float* __restrict__ whh_f, const float* __restrict__ whh_b,
                            const float* __restrict__ bih_f, const float* __restrict__ bhh_f,
                            const float* __restrict__ bih_b, const float* __restrict__ bhh_b,
                            const float* __restrict__ h0_f, const float* __restrict__ c0_f,
                            const float* __restrict__ h0_b, const float* __restrict__ c0_b,
                            float* __restrict__ catout, int Hdim, int catBase) {
  int dir = blockIdx.x;
  const float* gx  = dir ? gx_b  : gx_f;
  const float* whh = dir ? whh_b : whh_f;
  const float* bih = dir ? bih_b : bih_f;
  const float* bhh = dir ? bhh_b : bhh_f;
  const float* h0  = dir ? h0_b  : h0_f;
  const float* c0  = dir ? c0_b  : c0_f;
  int j = threadIdx.x;
  __shared__ float h_sh[H_W];
  int G = 4 * Hdim;
  float c = 0.0f;
  if (j < Hdim) { c = c0[j]; h_sh[j] = h0[j]; }
  __syncthreads();
  const float* wi = whh + (size_t)(0 * Hdim + j) * Hdim;
  const float* wf = whh + (size_t)(1 * Hdim + j) * Hdim;
  const float* wg = whh + (size_t)(2 * Hdim + j) * Hdim;
  const float* wo = whh + (size_t)(3 * Hdim + j) * Hdim;
  int off = catBase + (dir ? Hdim : 0);
  for (int step = 0; step < S_LEN; ++step) {
    int t = dir ? (S_LEN - 1 - step) : step;
    float hnew = 0.0f;
    if (j < Hdim) {
      float ai = 0.f, af = 0.f, ag = 0.f, ao = 0.f;
      for (int k = 0; k < Hdim; ++k) {
        float hk = h_sh[k];
        ai += wi[k] * hk; af += wf[k] * hk; ag += wg[k] * hk; ao += wo[k] * hk;
      }
      const float* gxt = gx + (size_t)t * G;
      float gi = sigmoidf_(gxt[j]          + bih[j]          + bhh[j]          + ai);
      float gf = sigmoidf_(gxt[Hdim + j]   + bih[Hdim + j]   + bhh[Hdim + j]   + af);
      float gg = tanhf    (gxt[2*Hdim + j] + bih[2*Hdim + j] + bhh[2*Hdim + j] + ag);
      float go = sigmoidf_(gxt[3*Hdim + j] + bih[3*Hdim + j] + bhh[3*Hdim + j] + ao);
      c = gf * c + gi * gg;
      hnew = go * tanhf(c);
      catout[(size_t)t * CAT_DIM + off + j] = hnew;
    }
    __syncthreads();
    if (j < Hdim) h_sh[j] = hnew;
    __syncthreads();
  }
}

// ---------------- CRF log-partition (single wave) ----------------
__global__ void k_crf(const float* __restrict__ feats, const float* __restrict__ trans,
                      float* __restrict__ out) {
  int n = threadIdx.x;
  __shared__ float fv[T_TAG];
  if (n < T_TAG) fv[n] = (n == START_IX) ? 0.0f : NEGVAL;
  __syncthreads();
  for (int t = 0; t < S_LEN; ++t) {
    float nv = 0.0f;
    if (n < T_TAG) {
      const float* tr = trans + n * T_TAG;
      float m = -3.4e38f;
      for (int p = 0; p < T_TAG; ++p) m = fmaxf(m, fv[p] + tr[p]);
      float s = 0.0f;
      for (int p = 0; p < T_TAG; ++p) s += __expf(fv[p] + tr[p] - m);
      nv = m + __logf(s) + feats[t * TPAD + n];
    }
    __syncthreads();
    if (n < T_TAG) fv[n] = nv;
    __syncthreads();
  }
  if (n == 0) {
    const float* tr = trans + STOP_IX * T_TAG;
    float m = -3.4e38f;
    for (int p = 0; p < T_TAG; ++p) m = fmaxf(m, fv[p] + tr[p]);
    float s = 0.0f;
    for (int p = 0; p < T_TAG; ++p) s += __expf(fv[p] + tr[p] - m);
    out[0] = m + __logf(s);
  }
}

// ---------------- host launcher ----------------
static inline int cdiv(int a, int b) { return (a + b - 1) / b; }

extern "C" void kernel_launch(void* const* d_in, const int* in_sizes, int n_in,
                              void* d_out, int out_size, void* d_ws, size_t ws_size,
                              hipStream_t stream) {
  const float* word_emb   = (const float*)d_in[0];
  const float* char_emb   = (const float*)d_in[1];
  const float* conv_w     = (const float*)d_in[2];
  const float* conv_b     = (const float*)d_in[3];
  const float* chw_fc1_w  = (const float*)d_in[4];
  const float* chw_fc1_b  = (const float*)d_in[5];
  const float* chw_gate_w = (const float*)d_in[6];
  const float* chw_gate_b = (const float*)d_in[7];
  const float* chw_lin_w  = (const float*)d_in[8];
  const float* chw_lin_b  = (const float*)d_in[9];
  const float* hw_fc1_w   = (const float*)d_in[10];
  const float* hw_fc1_b   = (const float*)d_in[11];
  const float* hw_gate_w  = (const float*)d_in[12];
  const float* hw_gate_b  = (const float*)d_in[13];
  const float* hw_lin_w   = (const float*)d_in[14];
  const float* hw_lin_b   = (const float*)d_in[15];
  const float* h2t_w      = (const float*)d_in[16];
  const float* h2t_b      = (const float*)d_in[17];
  const float* wf_Wih = (const float*)d_in[18];
  const float* wf_Whh = (const float*)d_in[19];
  const float* wf_bih = (const float*)d_in[20];
  const float* wf_bhh = (const float*)d_in[21];
  const float* wf_h0  = (const float*)d_in[22];
  const float* wf_c0  = (const float*)d_in[23];
  const float* wb_Wih = (const float*)d_in[24];
  const float* wb_Whh = (const float*)d_in[25];
  const float* wb_bih = (const float*)d_in[26];
  const float* wb_bhh = (const float*)d_in[27];
  const float* wb_h0  = (const float*)d_in[28];
  const float* wb_c0  = (const float*)d_in[29];
  const float* cf_Wih = (const float*)d_in[30];
  const float* cf_Whh = (const float*)d_in[31];
  const float* cf_bih = (const float*)d_in[32];
  const float* cf_bhh = (const float*)d_in[33];
  const float* cf_h0  = (const float*)d_in[34];
  const float* cf_c0  = (const float*)d_in[35];
  const float* cb_Wih = (const float*)d_in[36];
  const float* cb_Whh = (const float*)d_in[37];
  const float* cb_bih = (const float*)d_in[38];
  const float* cb_bhh = (const float*)d_in[39];
  const float* cb_h0  = (const float*)d_in[40];
  const float* cb_c0  = (const float*)d_in[41];
  const float* trans  = (const float*)d_in[42];
  const int* sentence = (const int*)d_in[43];
  const int* chars2   = (const int*)d_in[44];
  (void)in_sizes; (void)n_in; (void)out_size; (void)ws_size;

  char* base = (char*)d_ws;
  size_t off = 0;
  auto alloc = [&](size_t bytes) -> void* {
    off = (off + 255) & ~(size_t)255;
    void* p = base + off;
    off += bytes;
    return p;
  };

  unsigned short* we_bf    = (unsigned short*)alloc((size_t)S_LEN * E_DIM * 2);
  unsigned short* wihwf_bf = (unsigned short*)alloc((size_t)4 * H_W * E_DIM * 2);
  unsigned short* wihwb_bf = (unsigned short*)alloc((size_t)4 * H_W * E_DIM * 2);
  unsigned short* wihcf_bf = (unsigned short*)alloc((size_t)4 * H_C * OC_PAD * 2);
  unsigned short* wihcb_bf = (unsigned short*)alloc((size_t)4 * H_C * OC_PAD * 2);
  unsigned short* fc1w_bf  = (unsigned short*)alloc((size_t)CAT_DIM * CAT_DIM * 2);
  unsigned short* gatew_bf = (unsigned short*)alloc((size_t)CAT_DIM * CAT_DIM * 2);
  unsigned short* linw_bf  = (unsigned short*)alloc((size_t)HALF_HP * CAT_DIM * 2);
  unsigned short* h2tw_bf  = (unsigned short*)alloc((size_t)TPAD * HALF_HP * 2);
  float* linb_pad  = (float*)alloc(HALF_HP * 4);
  float* h2tb_pad  = (float*)alloc(TPAD * 4);
  float* cfeat     = (float*)alloc((size_t)S_LEN * OC_DIM * 4);
  float* cflow     = (float*)alloc((size_t)S_LEN * OC_DIM * 4);
  float* chw       = (float*)alloc((size_t)S_LEN * OC_DIM * 4);
  unsigned short* cfeat_bf = (unsigned short*)alloc((size_t)S_LEN * OC_PAD * 2);
  float* gx_wf     = (float*)alloc((size_t)S_LEN * 4 * H_W * 4);   // reused as hwh
  float* gx_wb     = (float*)alloc((size_t)S_LEN * 4 * H_W * 4);   // reused as hwt
  float* gx_cf     = (float*)alloc((size_t)S_LEN * 4 * H_C * 4);   // reused as linout
  float* gx_cb     = (float*)alloc((size_t)S_LEN * 4 * H_C * 4);   // reused as lin_bf
  float* cat       = (float*)alloc((size_t)S_LEN * CAT_DIM * 4);
  unsigned short* cat_bf  = (unsigned short*)alloc((size_t)S_LEN * CAT_DIM * 2);
  unsigned short* flow_bf = (unsigned short*)alloc((size_t)S_LEN * CAT_DIM * 2);
  float* feats     = (float*)alloc((size_t)S_LEN * TPAD * 4);
  float* hwh = gx_wf;
  float* hwt = gx_wb;
  float* linout = gx_cf;
  unsigned short* lin_bf = (unsigned short*)gx_cb;

  const int TB = 256;
  int n;
  n = 4 * H_W * E_DIM;
  k_convert_bf16<<<cdiv(n, TB), TB, 0, stream>>>(wf_Wih, wihwf_bf, n);
  k_convert_bf16<<<cdiv(n, TB), TB, 0, stream>>>(wb_Wih, wihwb_bf, n);
  n = 4 * H_C * OC_PAD;
  k_convert_pad_bf16<<<cdiv(n, TB), TB, 0, stream>>>(cf_Wih, wihcf_bf, 4 * H_C, OC_DIM, 4 * H_C, OC_PAD);
  k_convert_pad_bf16<<<cdiv(n, TB), TB, 0, stream>>>(cb_Wih, wihcb_bf, 4 * H_C, OC_DIM, 4 * H_C, OC_PAD);
  n = CAT_DIM * CAT_DIM;
  k_convert_bf16<<<cdiv(n, TB), TB, 0, stream>>>(hw_fc1_w, fc1w_bf, n);
  k_convert_bf16<<<cdiv(n, TB), TB, 0, stream>>>(hw_gate_w, gatew_bf, n);
  n = HALF_HP * CAT_DIM;
  k_convert_pad_bf16<<<cdiv(n, TB), TB, 0, stream>>>(hw_lin_w, linw_bf, HALF_H, CAT_DIM, HALF_HP, CAT_DIM);
  n = TPAD * HALF_HP;
  k_convert_pad_bf16<<<cdiv(n, TB), TB, 0, stream>>>(h2t_w, h2tw_bf, T_TAG, HALF_H, TPAD, HALF_HP);
  k_pad_f32<<<cdiv(HALF_HP, TB), TB, 0, stream>>>(hw_lin_b, linb_pad, HALF_H, HALF_HP);
  k_pad_f32<<<1, TPAD, 0, stream>>>(h2t_b, h2tb_pad, T_TAG, TPAD);

  n = S_LEN * E_DIM;
  k_gather_emb<<<cdiv(n, TB), TB, 0, stream>>>(word_emb, sentence, we_bf);
  n = S_LEN * OC_DIM;
  k_char_cnn<<<cdiv(n, TB), TB, 0, stream>>>(chars2, char_emb, conv_w, conv_b, cfeat);
  k_char_hw_flow<<<cdiv(n, TB), TB, 0, stream>>>(cfeat, chw_fc1_w, chw_fc1_b, chw_gate_w, chw_gate_b, cflow);
  k_char_hw_lin<<<cdiv(n, TB), TB, 0, stream>>>(cflow, chw_lin_w, chw_lin_b, chw);
  n = S_LEN * OC_PAD;
  k_convert_pad_bf16<<<cdiv(n, TB), TB, 0, stream>>>(chw, cfeat_bf, S_LEN, OC_DIM, S_LEN, OC_PAD);

  auto gemm = [&](const unsigned short* Am, const unsigned short* Wm, const float* bias,
                  float* Cm, int N_, int K_) {
    dim3 g(N_ / 16, S_LEN / 256);
    k_gemm_bf16_wmma<<<g, 128, 0, stream>>>(Am, Wm, bias, Cm, N_, K_);
  };
  gemm(we_bf,    wihwf_bf, nullptr, gx_wf, 4 * H_W, E_DIM);
  gemm(we_bf,    wihwb_bf, nullptr, gx_wb, 4 * H_W, E_DIM);
  gemm(cfeat_bf, wihcf_bf, nullptr, gx_cf, 4 * H_C, OC_PAD);
  gemm(cfeat_bf, wihcb_bf, nullptr, gx_cb, 4 * H_C, OC_PAD);

  k_lstm_scan<<<2, 1024, 0, stream>>>(gx_wf, gx_wb, wf_Whh, wb_Whh, wf_bih, wf_bhh, wb_bih, wb_bhh,
                                      wf_h0, wf_c0, wb_h0, wb_c0, cat, H_W, 0);
  k_lstm_scan<<<2, 256, 0, stream>>>(gx_cf, gx_cb, cf_Whh, cb_Whh, cf_bih, cf_bhh, cb_bih, cb_bhh,
                                     cf_h0, cf_c0, cb_h0, cb_c0, cat, H_C, 2 * H_W);

  n = S_LEN * CAT_DIM;
  k_convert_bf16<<<cdiv(n, TB), TB, 0, stream>>>(cat, cat_bf, n);
  gemm(cat_bf, fc1w_bf,  hw_fc1_b,  hwh, CAT_DIM, CAT_DIM);
  gemm(cat_bf, gatew_bf, hw_gate_b, hwt, CAT_DIM, CAT_DIM);
  k_hw_fuse<<<cdiv(n, TB), TB, 0, stream>>>(hwh, hwt, cat, flow_bf, n);
  gemm(flow_bf, linw_bf, linb_pad, linout, HALF_HP, CAT_DIM);
  n = S_LEN * HALF_HP;
  k_convert_bf16<<<cdiv(n, TB), TB, 0, stream>>>(linout, lin_bf, n);
  gemm(lin_bf, h2tw_bf, h2tb_pad, feats, TPAD, HALF_HP);

  k_crf<<<1, 32, 0, stream>>>(feats, trans, (float*)d_out);
}
